// CausalSelfAttention_13872744366131
// MI455X (gfx1250) — matrix operand order repaired and checked by reading, not compiled, not run on hardware
//
#include <hip/hip_runtime.h>

// ---------------------------------------------------------------------------
// CausalSelfAttention for MI455X (gfx1250, wave32, WMMA bf16 16x16x32)
// x[4,2048,1024] fp32, W_qkv[3072,1024], W_proj[1024,1024] -> out fp32
// Pipeline: cvt->bf16, WMMA GEMM with double-buffered ASYNC global->LDS
// staging (qkv epilogue scatters Q,K [B,H,T,64] and V transposed [B,H,64,T]),
// flash-attention (WMMA QK^T + PV), WMMA GEMM proj.
// ---------------------------------------------------------------------------

typedef __attribute__((ext_vector_type(16))) __bf16 bf16x16;
typedef __attribute__((ext_vector_type(8)))  __bf16 bf16x8;
typedef __attribute__((ext_vector_type(4)))  __bf16 bf16x4;
typedef __attribute__((ext_vector_type(8)))  float  f32x8;
typedef __attribute__((ext_vector_type(4)))  float  f32x4;

#define NB   4
#define NT   2048
#define NC   1024
#define NH   16
#define ND   64

union BF16V { bf16x16 v; bf16x8 h[2]; };

__device__ __forceinline__ f32x8 wmma_bf16(bf16x16 a, bf16x16 b, f32x8 c) {
  // D = A(16x32) * B(32x16) + C, fp32 accumulate
  return __builtin_amdgcn_wmma_f32_16x16x32_bf16(false, a, false, b,
                                                 (short)0, c, false, false);
}

// Async 16B global->LDS copy (CDNA5 GLOBAL_LOAD_ASYNC_TO_LDS_B128, GV mode).
// No data VGPRs; tracked by ASYNCcnt. LDS dest = low 32 bits of flat pointer.
__device__ __forceinline__ void async_cp16(const __bf16* g, __bf16* l) {
  uint32_t ldst = (uint32_t)(uintptr_t)l;
  uint64_t gsrc = (uint64_t)(uintptr_t)g;
  asm volatile("global_load_async_to_lds_b128 %0, %1, off"
               :: "v"(ldst), "v"(gsrc) : "memory");
}

__device__ __forceinline__ void wait_async0() {
  asm volatile("s_wait_asynccnt 0" ::: "memory");
}

// A fragment (16x32 bf16, row-major source, stride ld elements):
// lane m=lane&15 holds row m; kh=lane>>4; VGPR0-3: K=kh*8+0..7, VGPR4-7: K=16+kh*8+0..7
__device__ __forceinline__ bf16x16 load_frag_a(const __bf16* p, int ld) {
  const int lane = threadIdx.x & 31;
  const __bf16* r = p + (lane & 15) * ld + ((lane >> 4) << 3);
  BF16V u;
  u.h[0] = *(const bf16x8*)(r);
  u.h[1] = *(const bf16x8*)(r + 16);
  return u.v;
}

// B fragment (32x16 logical K x N). Source is "row-major second operand":
// lane n=lane&15 holds source row n (= output column n), K = kh*8+0..7, 16+kh*8+0..7.
__device__ __forceinline__ bf16x16 load_frag_b(const __bf16* p, int ld) {
  const int lane = threadIdx.x & 31;
  const __bf16* r = p + (lane & 15) * ld + ((lane >> 4) << 3);
  BF16V u;
  u.h[0] = *(const bf16x8*)(r);
  u.h[1] = *(const bf16x8*)(r + 16);
  return u.v;
}

// ---------------------------------------------------------------------------
__global__ __launch_bounds__(256) void cvt_f32_to_bf16(
    const float* __restrict__ in, __bf16* __restrict__ out, int n4) {
  int i = blockIdx.x * blockDim.x + threadIdx.x;
  if (i < n4) {
    f32x4 v = *(const f32x4*)(in + (size_t)i * 4);
    bf16x4 o;
    o.x = (__bf16)v.x; o.y = (__bf16)v.y; o.z = (__bf16)v.z; o.w = (__bf16)v.w;
    *(bf16x4*)(out + (size_t)i * 4) = o;
  }
}

// ---------------------------------------------------------------------------
// C[M,N] = A[M,K] * W[N,K]^T, bf16 in, fp32 accumulate.
// Block: 256 thr (8 waves) -> 128x256 tile; wave -> 64x64 (4x4 WMMA frags).
// Double-buffered LDS, async global->LDS staging overlapped with WMMA.
// MODE 0: scatter to Q,K [B,H,T,64] + V^T [B,H,64,T] (bf16)
// MODE 1: write fp32 Out[M,N]
template<int MODE>
__global__ __launch_bounds__(256) void gemm_bf16_k(
    const __bf16* __restrict__ A, const __bf16* __restrict__ W,
    int M, int N, int K,
    __bf16* __restrict__ Qb, __bf16* __restrict__ Kb, __bf16* __restrict__ Vt,
    float* __restrict__ Out)
{
  __shared__ __bf16 sA[2][128 * 40];   // 128x32 per buffer, padded stride 40
  __shared__ __bf16 sB[2][256 * 40];   // 256x32 per buffer, padded stride 40

  const int tid = threadIdx.x;
  const int M0  = blockIdx.y * 128;
  const int N0  = blockIdx.x * 256;
  const int w   = tid >> 5;
  const int wm  = (w & 1) * 64;
  const int wn  = (w >> 1) * 64;

  // issue async DMA for one 32-wide K slice into LDS buffer `buf`
  auto stage = [&](int buf, int kt) {
    #pragma unroll
    for (int i = 0; i < 2; ++i) {          // A: 512 x 16B chunks
      int c = tid + i * 256;
      int row = c >> 2, col = (c & 3) << 3;
      async_cp16(&A[(size_t)(M0 + row) * K + kt + col], &sA[buf][row * 40 + col]);
    }
    #pragma unroll
    for (int i = 0; i < 4; ++i) {          // B: 1024 x 16B chunks
      int c = tid + i * 256;
      int row = c >> 2, col = (c & 3) << 3;
      async_cp16(&W[(size_t)(N0 + row) * K + kt + col], &sB[buf][row * 40 + col]);
    }
  };

  const f32x8 z = {0.f,0.f,0.f,0.f,0.f,0.f,0.f,0.f};
  f32x8 acc[4][4];
  #pragma unroll
  for (int i = 0; i < 4; ++i)
    #pragma unroll
    for (int jn = 0; jn < 4; ++jn) acc[i][jn] = z;

  stage(0, 0);
  wait_async0();
  __syncthreads();

  for (int kt = 0; kt < K; kt += 32) {
    const int cur = (kt >> 5) & 1;
    if (kt + 32 < K) stage(cur ^ 1, kt + 32);   // prefetch next slice (async)

    bf16x16 af[4], bfm[4];
    #pragma unroll
    for (int i = 0; i < 4; ++i)
      af[i] = load_frag_a(&sA[cur][(wm + i * 16) * 40], 40);
    #pragma unroll
    for (int jn = 0; jn < 4; ++jn)
      bfm[jn] = load_frag_b(&sB[cur][(wn + jn * 16) * 40], 40);
    #pragma unroll
    for (int i = 0; i < 4; ++i)
      #pragma unroll
      for (int jn = 0; jn < 4; ++jn)
        acc[i][jn] = wmma_bf16(af[i], bfm[jn], acc[i][jn]);

    wait_async0();        // next-slice DMA landed (this wave)
    __syncthreads();      // all waves done reading `cur`, all DMA visible
  }

  const int lane = tid & 31;
  const int nn = lane & 15, kh = lane >> 4;
  #pragma unroll
  for (int i = 0; i < 4; ++i) {
    #pragma unroll
    for (int jn = 0; jn < 4; ++jn) {
      #pragma unroll
      for (int r = 0; r < 8; ++r) {
        int row = M0 + wm + i * 16 + r + 8 * kh;
        int col = N0 + wn + jn * 16 + nn;
        float v = acc[i][jn][r];
        if (MODE == 0) {
          int sec = col >> 10;
          int cc  = col & 1023;
          int hh  = cc >> 6, dd = cc & 63;
          int bb  = row >> 11, tt = row & 2047;
          size_t bh = (size_t)(bb * NH + hh);
          if (sec == 0)      Qb[(bh * NT + tt) * ND + dd] = (__bf16)v;
          else if (sec == 1) Kb[(bh * NT + tt) * ND + dd] = (__bf16)v;
          else               Vt[(bh * ND + dd) * NT + tt] = (__bf16)v;   // transposed
        } else {
          Out[(size_t)row * N + col] = v;
        }
      }
    }
  }
}

// ---------------------------------------------------------------------------
// Flash attention: one wave per 16-query tile. D=64. Causal.
__global__ __launch_bounds__(256) void attn_k(
    const __bf16* __restrict__ Qb, const __bf16* __restrict__ Kb,
    const __bf16* __restrict__ Vt, __bf16* __restrict__ Yb)
{
  __shared__ __bf16 plds[8 * 16 * 40];   // 16x32 P tile per wave, stride 40

  const int tid  = threadIdx.x;
  const int w    = tid >> 5;
  const int lane = tid & 31;
  const int nn   = lane & 15, kh = lane >> 4;
  const int wid  = blockIdx.x * 8 + w;
  const int qt   = wid & 127;            // 128 query tiles of 16
  const int h    = (wid >> 7) & 15;
  const int b    = wid >> 11;
  const int q0   = qt * 16;

  const size_t bh = (size_t)(b * NH + h);
  const __bf16* Qh = Qb + bh * (size_t)NT * ND;
  const __bf16* Kh = Kb + bh * (size_t)NT * ND;
  const __bf16* Vh = Vt + bh * (size_t)ND * NT;
  __bf16* pw = &plds[w * 640];

  bf16x16 qa0 = load_frag_a(Qh + (size_t)q0 * ND, ND);        // d = 0..31
  bf16x16 qa1 = load_frag_a(Qh + (size_t)q0 * ND + 32, ND);   // d = 32..63

  const f32x8 z = {0.f,0.f,0.f,0.f,0.f,0.f,0.f,0.f};
  f32x8 o[4];
  #pragma unroll
  for (int t2 = 0; t2 < 4; ++t2) o[t2] = z;
  float mst[8], lst[8];
  #pragma unroll
  for (int r = 0; r < 8; ++r) { mst[r] = -3.0e38f; lst[r] = 0.f; }

  const float scale = 0.125f;   // 1/sqrt(64)

  for (int j = 0; j < q0 + 16; j += 32) {   // causal: skip fully-masked chunks
    f32x8 s0 = z, s1 = z;
    s0 = wmma_bf16(qa0, load_frag_b(Kh + (size_t)j * ND, ND), s0);
    s0 = wmma_bf16(qa1, load_frag_b(Kh + (size_t)j * ND + 32, ND), s0);
    s1 = wmma_bf16(qa0, load_frag_b(Kh + (size_t)(j + 16) * ND, ND), s1);
    s1 = wmma_bf16(qa1, load_frag_b(Kh + (size_t)(j + 16) * ND + 32, ND), s1);

    float corr[8];
    #pragma unroll
    for (int r = 0; r < 8; ++r) {
      int qrow = q0 + r + 8 * kh;
      float v0 = s0[r] * scale;
      float v1 = s1[r] * scale;
      if (j + nn > qrow)      v0 = -3.0e38f;   // causal mask
      if (j + 16 + nn > qrow) v1 = -3.0e38f;
      float cm = fmaxf(v0, v1);
      cm = fmaxf(cm, __shfl_xor(cm, 1, 32));
      cm = fmaxf(cm, __shfl_xor(cm, 2, 32));
      cm = fmaxf(cm, __shfl_xor(cm, 4, 32));
      cm = fmaxf(cm, __shfl_xor(cm, 8, 32));
      float mnew = fmaxf(mst[r], cm);
      float cr = __expf(mst[r] - mnew);
      float p0 = __expf(v0 - mnew);
      float p1 = __expf(v1 - mnew);
      float rs = p0 + p1;
      rs += __shfl_xor(rs, 1, 32);
      rs += __shfl_xor(rs, 2, 32);
      rs += __shfl_xor(rs, 4, 32);
      rs += __shfl_xor(rs, 8, 32);
      lst[r] = lst[r] * cr + rs;
      mst[r] = mnew;
      corr[r] = cr;
      // write P (C-layout -> row-major LDS) for A-fragment reload
      pw[(r + 8 * kh) * 40 + nn]      = (__bf16)p0;
      pw[(r + 8 * kh) * 40 + 16 + nn] = (__bf16)p1;
    }
    #pragma unroll
    for (int t2 = 0; t2 < 4; ++t2)
      #pragma unroll
      for (int r = 0; r < 8; ++r)
        o[t2][r] *= corr[r];

    bf16x16 pa = load_frag_a(pw, 40);   // same-wave LDS, DS ops are in-order
    #pragma unroll
    for (int t2 = 0; t2 < 4; ++t2)
      o[t2] = wmma_bf16(pa, load_frag_b(Vh + (size_t)(t2 * 16) * NT + j, NT), o[t2]);
  }

  float inv[8];
  #pragma unroll
  for (int r = 0; r < 8; ++r) inv[r] = 1.0f / lst[r];
  #pragma unroll
  for (int t2 = 0; t2 < 4; ++t2)
    #pragma unroll
    for (int r = 0; r < 8; ++r)
      Yb[((size_t)b * NT + q0 + r + 8 * kh) * NC + h * ND + t2 * 16 + nn] =
          (__bf16)(o[t2][r] * inv[r]);
}

// ---------------------------------------------------------------------------
extern "C" void kernel_launch(void* const* d_in, const int* in_sizes, int n_in,
                              void* d_out, int out_size, void* d_ws, size_t ws_size,
                              hipStream_t stream) {
  const float* x  = (const float*)d_in[0];
  const float* Wq = (const float*)d_in[1];
  const float* Wp = (const float*)d_in[2];
  float* out = (float*)d_out;

  const size_t XB_E  = (size_t)NB * NT * NC;        // 8388608
  const size_t WQ_E  = (size_t)3 * NC * NC;         // 3145728
  const size_t WP_E  = (size_t)NC * NC;             // 1048576
  const size_t QKV_E = (size_t)NB * NH * NT * ND;   // 8388608

  char* ws = (char*)d_ws;
  __bf16* Xb  = (__bf16*)ws; ws += XB_E * 2;
  __bf16* Wqb = (__bf16*)ws; ws += WQ_E * 2;
  __bf16* Wpb = (__bf16*)ws; ws += WP_E * 2;
  __bf16* Qb  = (__bf16*)ws; ws += QKV_E * 2;
  __bf16* Kb  = (__bf16*)ws; ws += QKV_E * 2;
  __bf16* Vt  = (__bf16*)ws; ws += QKV_E * 2;
  __bf16* Yb  = (__bf16*)ws; ws += XB_E * 2;
  // total workspace: ~92.3 MB

  cvt_f32_to_bf16<<<(int)(XB_E / 4 / 256), 256, 0, stream>>>(x,  Xb,  (int)(XB_E / 4));
  cvt_f32_to_bf16<<<(int)(WQ_E / 4 / 256), 256, 0, stream>>>(Wq, Wqb, (int)(WQ_E / 4));
  cvt_f32_to_bf16<<<(int)(WP_E / 4 / 256), 256, 0, stream>>>(Wp, Wpb, (int)(WP_E / 4));

  // QKV: [8192,1024] x [3072,1024]^T
  gemm_bf16_k<0><<<dim3(3072 / 256, 8192 / 128), 256, 0, stream>>>(
      Xb, Wqb, 8192, 3072, 1024, Qb, Kb, Vt, nullptr);

  // Attention: 8192 query-tile waves / 8 waves per block
  attn_k<<<1024, 256, 0, stream>>>(Qb, Kb, Vt, Yb);

  // Proj: [8192,1024] x [1024,1024]^T -> fp32 out
  gemm_bf16_k<1><<<dim3(1024 / 256, 8192 / 128), 256, 0, stream>>>(
      Yb, Wpb, 8192, 1024, 1024, nullptr, nullptr, nullptr, out);
}
